// FEDformer_44212393345568
// MI455X (gfx1250) — matrix operand order repaired and checked by reading, not compiled, not run on hardware
//
#include <hip/hip_runtime.h>
#include <math.h>

// ---------------- model dims ----------------
#define B_     16
#define LE_    720
#define PRED_  720
#define LABEL_ 360
#define LD_    1080   // LABEL + PRED
#define N_     321
#define D_     512
#define H_     8
#define E_     64
#define DFF_   2048
#define MODES_ 64
#define EL_    2

typedef __attribute__((ext_vector_type(16))) _Float16 v16h;
typedef __attribute__((ext_vector_type(8)))  _Float16 v8h;
typedef __attribute__((ext_vector_type(2)))  _Float16 v2h;
typedef __attribute__((ext_vector_type(8)))  float    v8f;
typedef __attribute__((ext_vector_type(2)))  float    v2f;

union frag16 { v16h v; v8h p[2]; };

// =====================================================================
// Generalized batched GEMM on the WMMA pipe:
//   C[bz][m*ldc + n] = act( alpha * sum_k A[bz][m*lam + k*lak] *
//                                         B[bz][n*lbn + k*lbk] + bias[n] )
// Block tile 64(M) x 128(N), 8 waves (4x2), 4 accumulators/wave,
// K-step 32, f16 inputs / f32 accumulate.
//  - row offsets/validity precomputed (loop-invariant)
//  - main K-chunks are clamp-free; single clamped tail chunk
//  - contiguous-K operands use 8-byte vector loads
//  - double-buffered LDS: one barrier/chunk, global loads overlap WMMA
// =====================================================================
__launch_bounds__(256)
__global__ void gemm_wmma_f16(const float* __restrict__ A, long long sAb, int lam, int lak,
                              const float* __restrict__ Bm, long long sBb, int lbn, int lbk,
                              const float* __restrict__ bias,
                              float* __restrict__ C, long long sCb, int ldc,
                              int M, int K, int N, int act, float alpha)
{
    __shared__ _Float16 As[2][64][40];    // 40-half rows: conflict-free, 16B-aligned rows
    __shared__ _Float16 Bs[2][128][40];

    const int tid  = threadIdx.x;
    const int lane = tid & 31;
    const int wv   = tid >> 5;              // 0..7
    const int wm   = (wv & 3) << 4;         // 0/16/32/48
    const int wn0  = (wv >> 2) << 6;        // 0/64
    const int bm   = blockIdx.y * 64;
    const int bn   = blockIdx.x * 128;
    const int bz   = blockIdx.z;

    const float* Ab = A  + (long long)bz * sAb;
    const float* Bb = Bm + (long long)bz * sBb;
    float*       Cb = C  + (long long)bz * sCb;

    // per-thread staging coordinates (loop-invariant)
    int ar_[4], ak_[4], avalid[4]; long long aoff[4];
    #pragma unroll
    for (int j = 0; j < 4; ++j) {
        int p = tid + (j << 8);
        ar_[j] = p >> 4; ak_[j] = (p & 15) << 1;
        int gm = bm + ar_[j];
        avalid[j] = gm < M;
        aoff[j] = (long long)(gm < M ? gm : M - 1) * lam;
    }
    int br_[8], bk_[8], bvalid[8]; long long boff[8];
    #pragma unroll
    for (int j = 0; j < 8; ++j) {
        int p = tid + (j << 8);
        br_[j] = p >> 4; bk_[j] = (p & 15) << 1;
        int gn = bn + br_[j];
        bvalid[j] = gn < N;
        boff[j] = (long long)(gn < N ? gn : N - 1) * lbn;
    }
    const bool avec = (lak == 1) && ((lam & 1) == 0);   // contiguous, 8B-aligned k-pairs
    const bool bvec = (lbk == 1) && ((lbn & 1) == 0);

    const int kmain  = K & ~31;
    const int nchunk = (K + 31) >> 5;

    float ax0[4], ax1[4], bx0[8], bx1[8];

    auto loadRegs = [&](int k0) {
        if (k0 < kmain) {                       // clamp-free main chunk
            #pragma unroll
            for (int j = 0; j < 4; ++j) {
                int gk = k0 + ak_[j];
                if (avec) { v2f t = *(const v2f*)(Ab + aoff[j] + gk); ax0[j] = t.x; ax1[j] = t.y; }
                else { ax0[j] = Ab[aoff[j] + (long long)gk * lak];
                       ax1[j] = Ab[aoff[j] + (long long)(gk + 1) * lak]; }
            }
            #pragma unroll
            for (int j = 0; j < 8; ++j) {
                int gk = k0 + bk_[j];
                if (bvec) { v2f t = *(const v2f*)(Bb + boff[j] + gk); bx0[j] = t.x; bx1[j] = t.y; }
                else { bx0[j] = Bb[boff[j] + (long long)gk * lbk];
                       bx1[j] = Bb[boff[j] + (long long)(gk + 1) * lbk]; }
            }
        } else {                                // clamped tail chunk
            #pragma unroll
            for (int j = 0; j < 4; ++j) {
                int gk = k0 + ak_[j];
                int g0 = gk < K ? gk : K - 1, g1 = (gk + 1) < K ? (gk + 1) : K - 1;
                ax0[j] = Ab[aoff[j] + (long long)g0 * lak];
                ax1[j] = Ab[aoff[j] + (long long)g1 * lak];
            }
            #pragma unroll
            for (int j = 0; j < 8; ++j) {
                int gk = k0 + bk_[j];
                int g0 = gk < K ? gk : K - 1, g1 = (gk + 1) < K ? (gk + 1) : K - 1;
                bx0[j] = Bb[boff[j] + (long long)g0 * lbk];
                bx1[j] = Bb[boff[j] + (long long)g1 * lbk];
            }
        }
    };

    auto storeLds = [&](int buf, int k0) {
        #pragma unroll
        for (int j = 0; j < 4; ++j) {
            int gk = k0 + ak_[j];
            v2h h;
            h.x = (_Float16)((avalid[j] && gk     < K) ? ax0[j] : 0.f);
            h.y = (_Float16)((avalid[j] && gk + 1 < K) ? ax1[j] : 0.f);
            *(v2h*)&As[buf][ar_[j]][ak_[j]] = h;
        }
        #pragma unroll
        for (int j = 0; j < 8; ++j) {
            int gk = k0 + bk_[j];
            v2h h;
            h.x = (_Float16)((bvalid[j] && gk     < K) ? bx0[j] : 0.f);
            h.y = (_Float16)((bvalid[j] && gk + 1 < K) ? bx1[j] : 0.f);
            *(v2h*)&Bs[buf][br_[j]][bk_[j]] = h;
        }
    };

    v8f acc[4];
    #pragma unroll
    for (int t = 0; t < 4; ++t) acc[t] = (v8f){0.f,0.f,0.f,0.f,0.f,0.f,0.f,0.f};

    // prologue: stage chunk 0
    loadRegs(0);
    storeLds(0, 0);
    __syncthreads();

    for (int c = 0; c < nchunk; ++c) {
        const int cur = c & 1;
        const int k0n = (c + 1) << 5;
        if (c + 1 < nchunk) loadRegs(k0n);      // overlap with WMMA below

        // prefetch 2 chunks ahead (global_prefetch_b8)
        if (c + 2 < nchunk) {
            int gk = ((c + 2) << 5) + ak_[0];
            int gkc = gk < K ? gk : K - 1;
            __builtin_prefetch(Ab + aoff[0] + (long long)gkc * lak, 0, 1);
            __builtin_prefetch(Bb + boff[0] + (long long)gkc * lbk, 0, 1);
        }

        // A fragment: lane holds M=lane%16, K = h%8 + (h/8)*16 + (lane/16)*8
        frag16 af;
        const int abase = (lane >> 4) << 3;          // 0 or 8
        const int arr   = wm + (lane & 15);
        af.p[0] = *(const v8h*)&As[cur][arr][abase];
        af.p[1] = *(const v8h*)&As[cur][arr][abase + 16];

        // B fragments: lane holds N=lane%16, K = h%8 + (h/8)*16 + (lane/16)*8
        const int kb = (lane >> 4) << 3;             // 0 or 8
        #pragma unroll
        for (int t = 0; t < 4; ++t) {
            frag16 bf;
            const int brr = wn0 + (t << 4) + (lane & 15);
            bf.p[0] = *(const v8h*)&Bs[cur][brr][kb];
            bf.p[1] = *(const v8h*)&Bs[cur][brr][kb + 16];
            acc[t] = __builtin_amdgcn_wmma_f32_16x16x32_f16(false, af.v, false, bf.v,
                                                            (short)0, acc[t], false, false);
        }

        if (c + 1 < nchunk) storeLds(1 - cur, k0n);
        __syncthreads();
    }

    // D layout: VGPR r -> M = r + 8*(lane/16), N = lane%16
    #pragma unroll
    for (int t = 0; t < 4; ++t) {
        int cn = bn + wn0 + (t << 4) + (lane & 15);
        if (cn < N) {
            float bval = bias ? bias[cn] : 0.f;
            #pragma unroll
            for (int r8 = 0; r8 < 8; ++r8) {
                int cm = bm + wm + r8 + ((lane >> 4) << 3);
                if (cm < M) {
                    float v = acc[t][r8] * alpha + bval;
                    if (act == 1) v = 0.5f * v * (1.0f + erff(v * 0.70710678118f)); // exact gelu
                    Cb[(long long)cm * ldc + cn] = v;
                }
            }
        }
    }
}

// =====================================================================
// Support kernels (VALU; ~16% of FLOPs)
// =====================================================================
__global__ void k_add(const float* __restrict__ a, const float* __restrict__ b,
                      float* __restrict__ c, long long n)
{
    long long i = blockIdx.x * 256LL + threadIdx.x;
    if (i < n) c[i] = a[i] + b[i];
}

// moving-average decomposition with edge-replication (k=25).
// mode: 0 = write trend, 1 = trend += ma, 2 = discard trend
__global__ void k_decomp(const float* __restrict__ x, float* __restrict__ s,
                         float* __restrict__ t, int L, int Cc, int mode, long long total)
{
    long long i = blockIdx.x * 256LL + threadIdx.x;
    if (i >= total) return;
    int c = (int)(i % Cc);
    long long r = i / Cc;
    int l = (int)(r % L);
    int b = (int)(r / L);
    const float* xb = x + (long long)b * L * Cc + c;
    float sum = 0.f;
    #pragma unroll 5
    for (int j = -12; j <= 12; ++j) {
        int ll = l + j;
        ll = ll < 0 ? 0 : (ll >= L ? L - 1 : ll);
        sum += xb[(long long)ll * Cc];
    }
    float ma = sum * (1.0f / 25.0f);
    s[i] = xb[(long long)l * Cc] - ma;
    if (mode == 0) t[i] = ma;
    else if (mode == 1) t[i] += ma;
}

// per-row layernorm over D (one 256-thread block per row)
__global__ void k_ln(const float* __restrict__ x, const float* __restrict__ g,
                     const float* __restrict__ bb, float* __restrict__ y, int Dm)
{
    __shared__ float red[256];
    long long row = blockIdx.x;
    const float* xr = x + row * Dm;
    float ls = 0.f;
    for (int j = threadIdx.x; j < Dm; j += 256) ls += xr[j];
    red[threadIdx.x] = ls; __syncthreads();
    for (int s = 128; s > 0; s >>= 1) { if (threadIdx.x < s) red[threadIdx.x] += red[threadIdx.x + s]; __syncthreads(); }
    float mu = red[0] / Dm; __syncthreads();
    float lv = 0.f;
    for (int j = threadIdx.x; j < Dm; j += 256) { float d = xr[j] - mu; lv += d * d; }
    red[threadIdx.x] = lv; __syncthreads();
    for (int s = 128; s > 0; s >>= 1) { if (threadIdx.x < s) red[threadIdx.x] += red[threadIdx.x + s]; __syncthreads(); }
    float inv = rsqrtf(red[0] / Dm + 1e-5f);
    for (int j = threadIdx.x; j < Dm; j += 256)
        y[row * Dm + j] = (xr[j] - mu) * inv * g[j] + bb[j];
}

__global__ void k_colmean(const float* __restrict__ y, float* __restrict__ cm,
                          int L, int Dm, long long total /* B*Dm */)
{
    long long i = blockIdx.x * 256LL + threadIdx.x;
    if (i >= total) return;
    int d = (int)(i % Dm); int b = (int)(i / Dm);
    const float* p = y + (long long)b * L * Dm + d;
    float s = 0.f;
    for (int l = 0; l < L; ++l) s += p[(long long)l * Dm];
    cm[i] = s / L;
}

__global__ void k_subcol(float* __restrict__ y, const float* __restrict__ cm,
                         int L, int Dm, long long total)
{
    long long i = blockIdx.x * 256LL + threadIdx.x;
    if (i >= total) return;
    int d = (int)(i % Dm);
    int b = (int)(i / ((long long)L * Dm));
    y[i] -= cm[(long long)b * Dm + d];
}

// DFT bases. fwd: [n(0..127)][t]  cos rows then -sin rows (rfft, e^{-i})
__global__ void k_basis_fwd(float* __restrict__ out, int L, long long total)
{
    long long i = blockIdx.x * 256LL + threadIdx.x;
    if (i >= total) return;
    int n = (int)(i / L), t = (int)(i % L);
    float ang = 6.28318530717958647692f * (float)((n & 63) * t) / (float)L;
    out[i] = (n < 64) ? cosf(ang) : -sinf(ang);
}
// inv: [t][k]  x[t] = sum_k ib[t,k]*X[k];  a0 = 1/L, am = 2/L
__global__ void k_basis_inv(float* __restrict__ out, int L, long long total)
{
    long long i = blockIdx.x * 256LL + threadIdx.x;
    if (i >= total) return;
    int t = (int)(i / 128), k = (int)(i % 128), m = k & 63;
    float ang = 6.28318530717958647692f * (float)(m * t) / (float)L;
    float a = ((m == 0) ? 1.0f : 2.0f) / (float)L;
    out[i] = (k < 64) ? a * cosf(ang) : -a * sinf(ang);
}

__global__ void k_meanN(const float* __restrict__ hist, float* __restrict__ mN, long long total)
{
    long long i = blockIdx.x * 256LL + threadIdx.x;
    if (i >= total) return;
    int n = (int)(i % N_), b = (int)(i / N_);
    float s = 0.f;
    for (int l = 0; l < LE_; ++l) s += hist[((long long)(b * LE_ + l) * N_ + n) * 5];
    mN[i] = s / (float)LE_;
}

// seasonal_init (zero-padded) and trend_init (trend tail + mean tile)
__global__ void k_inits(const float* __restrict__ hist, const float* __restrict__ mN,
                        float* __restrict__ s_init, float* __restrict__ t_init, long long total)
{
    long long i = blockIdx.x * 256LL + threadIdx.x;
    if (i >= total) return;
    int n = (int)(i % N_); long long r = i / N_;
    int l = (int)(r % LD_); int b = (int)(r / LD_);
    if (l < LABEL_) {
        int le = LE_ - LABEL_ + l;
        float sum = 0.f;
        for (int j = -12; j <= 12; ++j) {
            int ll = le + j; ll = ll < 0 ? 0 : (ll >= LE_ ? LE_ - 1 : ll);
            sum += hist[((long long)(b * LE_ + ll) * N_ + n) * 5];
        }
        float ma = sum * (1.0f / 25.0f);
        float xv = hist[((long long)(b * LE_ + le) * N_ + n) * 5];
        s_init[i] = xv - ma;
        t_init[i] = ma;
    } else {
        s_init[i] = 0.f;
        t_init[i] = mN[(long long)b * N_ + n];
    }
}

// circular conv (N->D, k=3) + time-feature embed, encoder input
__global__ void k_conv_enc(const float* __restrict__ hist, const float* __restrict__ Wv,
                           const float* __restrict__ Wt, float* __restrict__ out, long long total)
{
    long long i = blockIdx.x * 256LL + threadIdx.x;
    if (i >= total) return;
    int d = (int)(i % D_); long long r = i / D_;
    int l = (int)(r % LE_); int b = (int)(r / LE_);
    float acc = 0.f;
    for (int w = 0; w < 3; ++w) {
        int ls = (l - 1 + w + LE_) % LE_;
        const float* hp = hist + ((long long)(b * LE_ + ls) * N_) * 5;
        const float* wp = Wv + (long long)d * N_ * 3 + w;
        for (int n = 0; n < N_; ++n) acc += hp[n * 5] * wp[n * 3];
    }
    const float* mp = hist + ((long long)(b * LE_ + l) * N_) * 5 + 1;
    for (int j = 0; j < 4; ++j) acc += mp[j] * Wt[d * 4 + j];
    out[i] = acc;
}

// circular conv over seasonal_init + decoder time-feature embed
__global__ void k_conv_dec(const float* __restrict__ sinit, const float* __restrict__ hist,
                           const float* __restrict__ fut, const float* __restrict__ Wv,
                           const float* __restrict__ Wt, float* __restrict__ out, long long total)
{
    long long i = blockIdx.x * 256LL + threadIdx.x;
    if (i >= total) return;
    int d = (int)(i % D_); long long r = i / D_;
    int l = (int)(r % LD_); int b = (int)(r / LD_);
    float acc = 0.f;
    for (int w = 0; w < 3; ++w) {
        int ls = (l - 1 + w + LD_) % LD_;
        const float* xp = sinit + (long long)(b * LD_ + ls) * N_;
        const float* wp = Wv + (long long)d * N_ * 3 + w;
        for (int n = 0; n < N_; ++n) acc += xp[n] * wp[n * 3];
    }
    const float* mp = (l < LABEL_)
        ? hist + ((long long)(b * LE_ + (LE_ - LABEL_ + l)) * N_) * 5 + 1
        : fut  + ((long long)(b * PRED_ + (l - LABEL_)) * N_) * 5 + 1;
    for (int j = 0; j < 4; ++j) acc += mp[j] * Wt[d * 4 + j];
    out[i] = acc;
}

// trend = trend_init + circular conv (D->N, k=3) of summed trends
__global__ void k_conv_trend(const float* __restrict__ tsum, const float* __restrict__ Wtr,
                             const float* __restrict__ t_init, float* __restrict__ outT, long long total)
{
    long long i = blockIdx.x * 256LL + threadIdx.x;
    if (i >= total) return;
    int n = (int)(i % N_); long long r = i / N_;
    int l = (int)(r % LD_); int b = (int)(r / LD_);
    float acc = t_init[i];
    for (int w = 0; w < 3; ++w) {
        int ls = (l - 1 + w + LD_) % LD_;
        const float* xp = tsum + (long long)(b * LD_ + ls) * D_;
        const float* wp = Wtr + (long long)n * D_ * 3 + w;
        for (int d = 0; d < D_; ++d) acc += xp[d] * wp[d * 3];
    }
    outT[i] = acc;
}

// sel[b,h,o,m] = sum_e ft[b,h,e,m] * (fr + i*fi)[h,e,o,m]
// ftc/selc layout: [b][d=h*64+e][0:64]=re, [64:128]=im
__global__ void k_selmul(const float* __restrict__ ftc, const float* __restrict__ fr,
                         const float* __restrict__ fi, float* __restrict__ selc, long long total)
{
    long long i = blockIdx.x * 256LL + threadIdx.x;
    if (i >= total) return;                 // total = B*H*64*64
    int m = (int)(i & 63); long long r = i >> 6;
    int o = (int)(r & 63); r >>= 6;
    int h = (int)(r & 7);  int b = (int)(r >> 3);
    const float* fb  = ftc + (long long)(b * D_ + h * 64) * 128;
    const float* frb = fr + (long long)h * 64 * 64 * 64 + (long long)o * 64 + m;
    const float* fib = fi + (long long)h * 64 * 64 * 64 + (long long)o * 64 + m;
    float orr = 0.f, oii = 0.f;
    for (int e = 0; e < 64; ++e) {
        float xr = fb[e * 128 + m], xi = fb[e * 128 + 64 + m];
        float wr = frb[(long long)e * 4096], wi = fib[(long long)e * 4096];
        orr += xr * wr - xi * wi;
        oii += xr * wi + xi * wr;
    }
    long long ob = (long long)(b * D_ + h * 64 + o) * 128;
    selc[ob + m] = orr;
    selc[ob + 64 + m] = oii;
}

// a[b,h,x,y] = sum_e qf[b,h,e,x] * kf[b,h,e,y]
__global__ void k_attn_a(const float* __restrict__ fq, const float* __restrict__ fk,
                         float* __restrict__ fa, long long total)
{
    long long i = blockIdx.x * 256LL + threadIdx.x;
    if (i >= total) return;                 // total = B*H*64*64
    int y = (int)(i & 63); long long r = i >> 6;
    int x = (int)(r & 63); r >>= 6;
    int h = (int)(r & 7);  int b = (int)(r >> 3);
    const float* qb = fq + (long long)(b * D_ + h * 64) * 128;
    const float* kb = fk + (long long)(b * D_ + h * 64) * 128;
    float ar = 0.f, ai = 0.f;
    for (int e = 0; e < 64; ++e) {
        float qr = qb[e * 128 + x], qi = qb[e * 128 + 64 + x];
        float kr = kb[e * 128 + y], ki = kb[e * 128 + 64 + y];
        ar += qr * kr - qi * ki;
        ai += qr * ki + qi * kr;
    }
    long long ob = (long long)((b * H_ + h) * 64 + x) * 128;
    fa[ob + y] = ar;
    fa[ob + 64 + y] = ai;
}

// complex tanh: tanh(x+iy) = (sinh2x + i*sin2y)/(cosh2x + cos2y)
__global__ void k_ctanh(float* __restrict__ fa, long long total /* B*H*64*64 */)
{
    long long i = blockIdx.x * 256LL + threadIdx.x;
    if (i >= total) return;
    long long base = (i >> 6) * 128 + (i & 63);
    float x = fa[base], y = fa[base + 64];
    float den = coshf(2.f * x) + cosf(2.f * y);
    fa[base]      = sinhf(2.f * x) / den;
    fa[base + 64] = sinf(2.f * y) / den;
}

// v[b,h,e,x] = sum_y a[b,h,x,y] * kf[b,h,e,y]
__global__ void k_attn_v(const float* __restrict__ fa, const float* __restrict__ fk,
                         float* __restrict__ fv, long long total)
{
    long long i = blockIdx.x * 256LL + threadIdx.x;
    if (i >= total) return;                 // total = B*D*64
    int x = (int)(i & 63); long long r = i >> 6;
    int e = (int)(r & 63); r >>= 6;
    int h = (int)(r & 7);  int b = (int)(r >> 3);
    const float* ab = fa + (long long)((b * H_ + h) * 64 + x) * 128;
    const float* kb = fk + (long long)(b * D_ + h * 64 + e) * 128;
    float vr = 0.f, vi = 0.f;
    for (int y = 0; y < 64; ++y) {
        float ar = ab[y], ai = ab[64 + y];
        float kr = kb[y], ki = kb[64 + y];
        vr += ar * kr - ai * ki;
        vi += ar * ki + ai * kr;
    }
    long long ob = (long long)(b * D_ + h * 64 + e) * 128;
    fv[ob + x] = vr;
    fv[ob + 64 + x] = vi;
}

__global__ void k_final(const float* __restrict__ trendF, const float* __restrict__ seas,
                        float* __restrict__ out, long long total)
{
    long long i = blockIdx.x * 256LL + threadIdx.x;
    if (i >= total) return;                 // total = B*PRED*N
    int n = (int)(i % N_); long long r = i / N_;
    int p = (int)(r % PRED_); int b = (int)(r / PRED_);
    long long src = (long long)(b * LD_ + LABEL_ + p) * N_ + n;
    out[i] = trendF[src] + seas[src];
}

// =====================================================================
// Host orchestration
// =====================================================================
extern "C" void kernel_launch(void* const* d_in, const int* in_sizes, int n_in,
                              void* d_out, int out_size, void* d_ws, size_t ws_size,
                              hipStream_t stream)
{
    const float* hist   = (const float*)d_in[0];
    const float* fut    = (const float*)d_in[1];
    const float* Wv_enc = (const float*)d_in[2];
    const float* Wt_enc = (const float*)d_in[3];
    const float* Wv_dec = (const float*)d_in[4];
    const float* Wt_dec = (const float*)d_in[5];
    const float* eWq    = (const float*)d_in[6];
    const float* ebq    = (const float*)d_in[7];
    const float* efr    = (const float*)d_in[8];
    const float* efi    = (const float*)d_in[9];
    const float* eWo    = (const float*)d_in[10];
    const float* ebo    = (const float*)d_in[11];
    const float* eW1    = (const float*)d_in[12];
    const float* eW2    = (const float*)d_in[13];
    const float* g_enc  = (const float*)d_in[14];
    const float* b_enc  = (const float*)d_in[15];
    const float* dWq    = (const float*)d_in[16];
    const float* dbq    = (const float*)d_in[17];
    const float* dfr    = (const float*)d_in[18];
    const float* dfi    = (const float*)d_in[19];
    const float* dWo    = (const float*)d_in[20];
    const float* dbo    = (const float*)d_in[21];
    const float* cWq    = (const float*)d_in[22];
    const float* cbq    = (const float*)d_in[23];
    const float* cWk    = (const float*)d_in[24];
    const float* cbk    = (const float*)d_in[25];
    const float* cfr    = (const float*)d_in[26];
    const float* cfi    = (const float*)d_in[27];
    const float* cWo    = (const float*)d_in[28];
    const float* cbo    = (const float*)d_in[29];
    const float* dW1    = (const float*)d_in[30];
    const float* dW2    = (const float*)d_in[31];
    const float* Wtr    = (const float*)d_in[32];
    const float* g_dec  = (const float*)d_in[33];
    const float* b_dec  = (const float*)d_in[34];
    const float* Wp     = (const float*)d_in[35];
    const float* bp     = (const float*)d_in[36];
    (void)in_sizes; (void)n_in; (void)out_size; (void)ws_size;

    // ---- workspace layout (floats) ----
    float* ws = (float*)d_ws;
    long long off = 0;
    auto alloc = [&](long long nf) { float* p = ws + off; off += nf; return p; };
    float* s_init  = alloc((long long)B_ * LD_ * N_);
    float* t_init  = alloc((long long)B_ * LD_ * N_);
    float* xe      = alloc((long long)B_ * LE_ * D_);
    float* enc_out = alloc((long long)B_ * LE_ * D_);
    float* xd      = alloc((long long)B_ * LD_ * D_);
    float* tmp1    = alloc((long long)B_ * LD_ * D_);
    float* tmp2    = alloc((long long)B_ * LD_ * D_);
    float* tsum    = alloc((long long)B_ * LD_ * D_);
    float* hid     = alloc((long long)B_ * LD_ * DFF_);
    float* ftq     = alloc((long long)B_ * D_ * 128);
    float* fsel    = alloc((long long)B_ * D_ * 128);
    float* fkc     = alloc((long long)B_ * D_ * 128);
    float* fa      = alloc((long long)B_ * H_ * 64 * 128);
    float* fvc     = alloc((long long)B_ * D_ * 128);
    float* cmb     = alloc((long long)B_ * D_);
    float* mN      = alloc((long long)B_ * N_);
    float* fwdE    = alloc((long long)128 * LE_);
    float* invE    = alloc((long long)LE_ * 128);
    float* fwdD    = alloc((long long)128 * LD_);
    float* invD    = alloc((long long)LD_ * 128);

    auto gemm = [&](const float* A, long long sAb, int lam, int lak,
                    const float* Bm, long long sBb, int lbn, int lbk,
                    const float* bias, float* C, long long sCb, int ldc,
                    int M, int K, int N, int batch, int act, float alpha) {
        dim3 g((N + 127) / 128, (M + 63) / 64, batch);
        gemm_wmma_f16<<<g, dim3(256), 0, stream>>>(A, sAb, lam, lak, Bm, sBb, lbn, lbk,
                                                   bias, C, sCb, ldc, M, K, N, act, alpha);
    };
    auto blocks = [](long long t) { return (int)((t + 255) / 256); };

    const long long nBLED = (long long)B_ * LE_ * D_;
    const long long nBLDD = (long long)B_ * LD_ * D_;
    const long long nSpec = (long long)B_ * H_ * 64 * 64;

    // ---- precompute ----
    k_basis_fwd<<<blocks(128LL * LE_), 256, 0, stream>>>(fwdE, LE_, 128LL * LE_);
    k_basis_inv<<<blocks((long long)LE_ * 128), 256, 0, stream>>>(invE, LE_, (long long)LE_ * 128);
    k_basis_fwd<<<blocks(128LL * LD_), 256, 0, stream>>>(fwdD, LD_, 128LL * LD_);
    k_basis_inv<<<blocks((long long)LD_ * 128), 256, 0, stream>>>(invD, LD_, (long long)LD_ * 128);
    k_meanN<<<blocks((long long)B_ * N_), 256, 0, stream>>>(hist, mN, (long long)B_ * N_);
    k_inits<<<blocks((long long)B_ * LD_ * N_), 256, 0, stream>>>(hist, mN, s_init, t_init,
                                                                  (long long)B_ * LD_ * N_);
    k_conv_enc<<<blocks(nBLED), 256, 0, stream>>>(hist, Wv_enc, Wt_enc, xe, nBLED);

    // ---- encoder layers ----
    for (int i = 0; i < EL_; ++i) {
        const float* Wq = eWq + (long long)i * D_ * D_;
        const float* bq = ebq + (long long)i * D_;
        const float* Wo = eWo + (long long)i * D_ * D_;
        const float* bo = ebo + (long long)i * D_;
        const float* fr = efr + (long long)i * H_ * E_ * E_ * MODES_;
        const float* fi = efi + (long long)i * H_ * E_ * E_ * MODES_;
        const float* W1 = eW1 + (long long)i * DFF_ * D_;
        const float* W2 = eW2 + (long long)i * D_ * DFF_;

        // q = x @ Wq^T + bq
        gemm(xe, 0, D_, 1, Wq, 0, D_, 1, bq, tmp1, 0, D_, B_ * LE_, D_, D_, 1, 0, 1.f);
        // rfft (first 64 modes) as DFT GEMM, per batch: [d x 128]
        gemm(tmp1, (long long)LE_ * D_, 1, D_, fwdE, 0, LE_, 1, nullptr,
             ftq, (long long)D_ * 128, 128, D_, LE_, 128, B_, 0, 1.f);
        // mode mixing
        k_selmul<<<blocks(nSpec), 256, 0, stream>>>(ftq, fr, fi, fsel, nSpec);
        // irfft as iDFT GEMM: [t x d]
        gemm(invE, 0, 128, 1, fsel, (long long)D_ * 128, 128, 1, nullptr,
             tmp2, (long long)LE_ * D_, D_, LE_, 128, D_, B_, 0, 1.f);
        // out proj, residual, decomp
        gemm(tmp2, 0, D_, 1, Wo, 0, D_, 1, bo, tmp1, 0, D_, B_ * LE_, D_, D_, 1, 0, 1.f);
        k_add<<<blocks(nBLED), 256, 0, stream>>>(xe, tmp1, tmp2, nBLED);
        k_decomp<<<blocks(nBLED), 256, 0, stream>>>(tmp2, xe, nullptr, LE_, D_, 2, nBLED);
        // FFN with fused gelu
        gemm(xe, 0, D_, 1, W1, 0, D_, 1, nullptr, hid, 0, DFF_, B_ * LE_, D_, DFF_, 1, 1, 1.f);
        gemm(hid, 0, DFF_, 1, W2, 0, DFF_, 1, nullptr, tmp1, 0, D_, B_ * LE_, DFF_, D_, 1, 0, 1.f);
        k_add<<<blocks(nBLED), 256, 0, stream>>>(xe, tmp1, tmp2, nBLED);
        k_decomp<<<blocks(nBLED), 256, 0, stream>>>(tmp2, xe, nullptr, LE_, D_, 2, nBLED);
    }
    // enc_out = myln(x)
    k_ln<<<B_ * LE_, 256, 0, stream>>>(xe, g_enc, b_enc, enc_out, D_);
    k_colmean<<<blocks((long long)B_ * D_), 256, 0, stream>>>(enc_out, cmb, LE_, D_, (long long)B_ * D_);
    k_subcol<<<blocks(nBLED), 256, 0, stream>>>(enc_out, cmb, LE_, D_, nBLED);

    // ---- decoder ----
    k_conv_dec<<<blocks(nBLDD), 256, 0, stream>>>(s_init, hist, fut, Wv_dec, Wt_dec, xd, nBLDD);

    // self fourier attention
    gemm(xd, 0, D_, 1, dWq, 0, D_, 1, dbq, tmp1, 0, D_, B_ * LD_, D_, D_, 1, 0, 1.f);
    gemm(tmp1, (long long)LD_ * D_, 1, D_, fwdD, 0, LD_, 1, nullptr,
         ftq, (long long)D_ * 128, 128, D_, LD_, 128, B_, 0, 1.f);
    k_selmul<<<blocks(nSpec), 256, 0, stream>>>(ftq, dfr, dfi, fsel, nSpec);
    gemm(invD, 0, 128, 1, fsel, (long long)D_ * 128, 128, 1, nullptr,
         tmp2, (long long)LD_ * D_, D_, LD_, 128, D_, B_, 0, 1.f);
    gemm(tmp2, 0, D_, 1, dWo, 0, D_, 1, dbo, tmp1, 0, D_, B_ * LD_, D_, D_, 1, 0, 1.f);
    k_add<<<blocks(nBLDD), 256, 0, stream>>>(xd, tmp1, tmp2, nBLDD);
    k_decomp<<<blocks(nBLDD), 256, 0, stream>>>(tmp2, xd, tsum, LD_, D_, 0, nBLDD);  // t1 (write)

    // cross fourier attention
    gemm(xd, 0, D_, 1, cWq, 0, D_, 1, cbq, tmp1, 0, D_, B_ * LD_, D_, D_, 1, 0, 1.f);
    gemm(tmp1, (long long)LD_ * D_, 1, D_, fwdD, 0, LD_, 1, nullptr,
         ftq, (long long)D_ * 128, 128, D_, LD_, 128, B_, 0, 1.f);
    gemm(enc_out, 0, D_, 1, cWk, 0, D_, 1, cbk, tmp2, 0, D_, B_ * LE_, D_, D_, 1, 0, 1.f);
    gemm(tmp2, (long long)LE_ * D_, 1, D_, fwdE, 0, LE_, 1, nullptr,
         fkc, (long long)D_ * 128, 128, D_, LE_, 128, B_, 0, 1.f);
    k_attn_a<<<blocks(nSpec), 256, 0, stream>>>(ftq, fkc, fa, nSpec);
    k_ctanh<<<blocks(nSpec), 256, 0, stream>>>(fa, nSpec);
    k_attn_v<<<blocks((long long)B_ * D_ * 64), 256, 0, stream>>>(fa, fkc, fvc, (long long)B_ * D_ * 64);
    k_selmul<<<blocks(nSpec), 256, 0, stream>>>(fvc, cfr, cfi, fsel, nSpec);
    gemm(invD, 0, 128, 1, fsel, (long long)D_ * 128, 128, 1, nullptr,
         tmp2, (long long)LD_ * D_, D_, LD_, 128, D_, B_, 0, 1.f / 262144.f); // /(D*D)
    gemm(tmp2, 0, D_, 1, cWo, 0, D_, 1, cbo, tmp1, 0, D_, B_ * LD_, D_, D_, 1, 0, 1.f);
    k_add<<<blocks(nBLDD), 256, 0, stream>>>(xd, tmp1, tmp2, nBLDD);
    k_decomp<<<blocks(nBLDD), 256, 0, stream>>>(tmp2, xd, tsum, LD_, D_, 1, nBLDD);  // + t2

    // decoder FFN
    gemm(xd, 0, D_, 1, dW1, 0, D_, 1, nullptr, hid, 0, DFF_, B_ * LD_, D_, DFF_, 1, 1, 1.f);
    gemm(hid, 0, DFF_, 1, dW2, 0, DFF_, 1, nullptr, tmp1, 0, D_, B_ * LD_, DFF_, D_, 1, 0, 1.f);
    k_add<<<blocks(nBLDD), 256, 0, stream>>>(xd, tmp1, tmp2, nBLDD);
    k_decomp<<<blocks(nBLDD), 256, 0, stream>>>(tmp2, xd, tsum, LD_, D_, 1, nBLDD);  // + t3

    // trend path: t_init + conv_circ(tsum, Wtr)  (s_init buffer free -> trendF)
    k_conv_trend<<<blocks((long long)B_ * LD_ * N_), 256, 0, stream>>>(tsum, Wtr, t_init, s_init,
                                                                       (long long)B_ * LD_ * N_);
    // seasonal path: myln(xd) @ Wp^T + bp  (t_init free after conv_trend)
    k_ln<<<B_ * LD_, 256, 0, stream>>>(xd, g_dec, b_dec, tmp1, D_);
    k_colmean<<<blocks((long long)B_ * D_), 256, 0, stream>>>(tmp1, cmb, LD_, D_, (long long)B_ * D_);
    k_subcol<<<blocks(nBLDD), 256, 0, stream>>>(tmp1, cmb, LD_, D_, nBLDD);
    gemm(tmp1, 0, D_, 1, Wp, 0, D_, 1, bp, t_init, 0, N_, B_ * LD_, D_, N_, 1, 0, 1.f);

    // final: (trend + seasonal)[:, -PRED:]
    k_final<<<blocks((long long)B_ * PRED_ * N_), 256, 0, stream>>>(s_init, t_init, (float*)d_out,
                                                                    (long long)B_ * PRED_ * N_);
}
